// RWKV_TimeMix_43525198578349
// MI455X (gfx1250) — compile-verified, matrix operand order checked by reading, not run on hardware
//
#include <hip/hip_runtime.h>
#include <hip/hip_bf16.h>

// ---------------------------------------------------------------------------
// RWKV TimeMix for MI455X (gfx1250, wave32, WMMA + TDM async tensor loads)
// B=4 T=2048 C=1024 H=16 D=64.
// GEMMs: V_WMMA_F32_16X16X32_F16, LDS-staged via TENSOR_LOAD_TO_LDS.
// WKV scan: chunk-parallel 3-pass linear-recurrence scan.
// ---------------------------------------------------------------------------

#define RB 4
#define RT 2048
#define RC 1024
#define RH 16
#define RD 64
#define RM (RB * RT)          // 8192 activation rows
#define NCH 32                // scan chunks along T
#define LCH (RT / NCH)        // 64 steps per chunk

typedef __attribute__((ext_vector_type(16))) _Float16 v16h;
typedef __attribute__((ext_vector_type(8)))  float    v8f;
typedef unsigned int v4u  __attribute__((ext_vector_type(4)));
typedef int          v8i_ __attribute__((ext_vector_type(8)));
typedef int          v4i_ __attribute__((ext_vector_type(4)));

#if defined(__has_builtin)
#  if __has_builtin(__builtin_amdgcn_tensor_load_to_lds) && \
      __has_builtin(__builtin_amdgcn_s_wait_tensorcnt)
#    define USE_TDM 1
#  endif
#endif
#ifndef USE_TDM
#  define USE_TDM 0
#endif

union FragU {
    v16h   v;
    float4 f4[2];
};

// Load one 16x32 f16 A/B fragment from a row-major buffer with leading dim
// `ld` (elements). CDNA5 layout: lanes 0-15 = rows 0-15 with K {0..7,16..23};
// lanes 16-31 = same rows with K {8..15,24..31}. Two 16-byte loads per lane.
__device__ __forceinline__ v16h load_frag16x32(const _Float16* __restrict__ base,
                                               int ld, int lane) {
    const int r    = lane & 15;
    const int koff = (lane >> 4) << 3;   // 0 or 8
    const _Float16* p = base + (size_t)r * ld + koff;
    FragU u;
    u.f4[0] = *reinterpret_cast<const float4*>(p);
    u.f4[1] = *reinterpret_cast<const float4*>(p + 16);
    return u.v;
}

// Store one 16x16 f32 C/D tile: VGPR rr -> M=rr (lanes 0-15) / M=rr+8 (16-31),
// lane&15 is N. Optional fused sigmoid.
__device__ __forceinline__ void store_tile16x16(const v8f acc,
                                                float* __restrict__ cbase,
                                                int ld, int lane, int act) {
    const int n     = lane & 15;
    const int mbase = (lane >> 4) << 3;
    float* p = cbase + (size_t)mbase * ld + n;
#pragma unroll
    for (int rr = 0; rr < 8; ++rr) {
        float val = acc[rr];
        if (act) val = 1.0f / (1.0f + __expf(-val));
        p[(size_t)rr * ld] = val;
    }
}

#define WMMA_F16(a, b, c) \
    __builtin_amdgcn_wmma_f32_16x16x32_f16(false, (a), false, (b), (short)0, (c), false, false)

#if USE_TDM
// ---- Tensor DMA descriptor builders (CDNA5 ISA) ------
// Group 0: [1:0]=count=1, [63:32]=lds_addr, [120:64]=global_addr, [127:126]=2.
__device__ __forceinline__ v4u tdm_g0(unsigned lds_addr, unsigned long long gaddr) {
    v4u g0;
    g0[0] = 1u;                                           // count=1, user desc
    g0[1] = lds_addr;
    g0[2] = (unsigned)(gaddr & 0xFFFFFFFFu);
    g0[3] = (unsigned)((gaddr >> 32) & 0x01FFFFFFu) | (2u << 30);  // type=2
    return g0;
}
// Group 1: data_size=1 (2B), tensor_dim0/1, tile_dim0/1, tensor_dim0_stride.
__device__ __forceinline__ v8i_ tdm_g1(unsigned tdim0, unsigned tdim1,
                                       unsigned tile0, unsigned tile1,
                                       unsigned long long stride0) {
    v8i_ g1;
    g1[0] = (int)(1u << 16);                              // data_size = 2 bytes
    g1[1] = (int)((tdim0 & 0xFFFFu) << 16);               // tensor_dim0 [63:48]
    g1[2] = (int)((tdim0 >> 16) | ((tdim1 & 0xFFFFu) << 16));
    g1[3] = (int)((tdim1 >> 16) | (tile0 << 16));         // tile_dim0 [127:112]
    g1[4] = (int)(tile1 & 0xFFFFu);                       // tile_dim1; tile_dim2=0
    g1[5] = (int)(stride0 & 0xFFFFFFFFull);               // dim0_stride lo32
    g1[6] = (int)((stride0 >> 32) & 0xFFFFull);           // dim0_stride hi16
    g1[7] = 0;
    return g1;
}
// Issue one 2-D tile DMA: rows x 64 f16 elements, row-major, into LDS.
// 6-arg builtin form: (g0, g1, g2, g3, g_ext, cpol); groups 2/3 unused for 2-D.
__device__ __forceinline__ void tdm_load_tile(const _Float16* gsrc, unsigned lds_off,
                                              unsigned tensor_rows, unsigned Kdim,
                                              unsigned tile_rows) {
    v4i_ z4 = {};
    v8i_ z8 = {};
    __builtin_amdgcn_tensor_load_to_lds(
        tdm_g0(lds_off, (unsigned long long)(uintptr_t)gsrc),
        tdm_g1(Kdim, tensor_rows, 64u, tile_rows, (unsigned long long)Kdim),
        z4, z4, z8, 0);
}
#endif

// ---------------------------------------------------------------------------
// Kernel 1: token shift + k/v/r mixes, fp32 -> f16
// ---------------------------------------------------------------------------
__global__ __launch_bounds__(256) void mix_kernel(
    const float* __restrict__ x,
    const float* __restrict__ mk, const float* __restrict__ mv,
    const float* __restrict__ mr,
    _Float16* __restrict__ xk, _Float16* __restrict__ xv,
    _Float16* __restrict__ xr) {
    const int idx = blockIdx.x * blockDim.x + threadIdx.x;
    if (idx >= RM * RC) return;
    const int c = idx & (RC - 1);
    const int t = (idx / RC) & (RT - 1);
    const float xc = x[idx];
    const float xp = (t == 0) ? 0.0f : x[idx - RC];
    const float k = mk[c], v = mv[c], r = mr[c];
    xk[idx] = (_Float16)(xc * k + xp * (1.0f - k));
    xv[idx] = (_Float16)(xc * v + xp * (1.0f - v));
    xr[idx] = (_Float16)(xc * r + xp * (1.0f - r));
}

// ---------------------------------------------------------------------------
// Kernel 2: fp32 -> f16 weight conversion
// ---------------------------------------------------------------------------
__global__ __launch_bounds__(256) void f32_to_f16_kernel(
    const float* __restrict__ src, _Float16* __restrict__ dst, int n) {
    const int idx = blockIdx.x * blockDim.x + threadIdx.x;
    if (idx < n) dst[idx] = (_Float16)src[idx];
}

// ---------------------------------------------------------------------------
// Kernel 3: GEMM  Out[m,n] = sum_k A[m,k] * W[n,k]   (A @ W^T)
// A: (Mdim x Kdim) f16 row-major, W: (Ndim x Kdim) f16 row-major.
// Block = 256 threads = 8 waves (2x4), wave tile 32x32, block tile 64x128.
// TDM path: double-buffered BK=64 stages DMA'd into LDS by wave 0, gated by
// s_wait_tensorcnt + barriers; fragments read via ds_load_b128.
// ---------------------------------------------------------------------------
__global__ __launch_bounds__(256) void gemm_wmma_kernel(
    const _Float16* __restrict__ A, const _Float16* __restrict__ W,
    float* __restrict__ Out, int Mdim, int Ndim, int Kdim, int act) {
    const int lane = threadIdx.x & 31;
    const int wave = threadIdx.x >> 5;
    const int wm   = wave & 1;
    const int wn   = wave >> 1;
    const int mBlk = blockIdx.y * 64;
    const int nBlk = blockIdx.x * 128;

    v8f acc00 = {}, acc01 = {}, acc10 = {}, acc11 = {};

#if USE_TDM
    __shared__ alignas(16) _Float16 ldsA[2][64 * 64];    //  8 KB x2
    __shared__ alignas(16) _Float16 ldsB[2][128 * 64];   // 16 KB x2

    const int nsteps = Kdim >> 6;                        // BK = 64
    const _Float16* Ab = A + (size_t)mBlk * Kdim;
    const _Float16* Wb = W + (size_t)nBlk * Kdim;

    if (wave == 0) {
        tdm_load_tile(Ab,      (unsigned)(uintptr_t)&ldsA[0][0], (unsigned)Mdim,
                      (unsigned)Kdim, 64u);
        tdm_load_tile(Wb,      (unsigned)(uintptr_t)&ldsB[0][0], (unsigned)Ndim,
                      (unsigned)Kdim, 128u);
        if (nsteps > 1) {
            tdm_load_tile(Ab + 64, (unsigned)(uintptr_t)&ldsA[1][0], (unsigned)Mdim,
                          (unsigned)Kdim, 64u);
            tdm_load_tile(Wb + 64, (unsigned)(uintptr_t)&ldsB[1][0], (unsigned)Ndim,
                          (unsigned)Kdim, 128u);
        }
    }

    for (int s = 0; s < nsteps; ++s) {
        if (wave == 0) {
            if (s + 1 < nsteps) __builtin_amdgcn_s_wait_tensorcnt((short)2);
            else                __builtin_amdgcn_s_wait_tensorcnt((short)0);
        }
        __syncthreads();

        const _Float16* At = &ldsA[s & 1][0];
        const _Float16* Bt = &ldsB[s & 1][0];
#pragma unroll
        for (int kk = 0; kk < 64; kk += 32) {
            const v16h a0 = load_frag16x32(At + (size_t)(wm * 32)      * 64 + kk, 64, lane);
            const v16h a1 = load_frag16x32(At + (size_t)(wm * 32 + 16) * 64 + kk, 64, lane);
            const v16h b0 = load_frag16x32(Bt + (size_t)(wn * 32)      * 64 + kk, 64, lane);
            const v16h b1 = load_frag16x32(Bt + (size_t)(wn * 32 + 16) * 64 + kk, 64, lane);
            acc00 = WMMA_F16(a0, b0, acc00);
            acc01 = WMMA_F16(a0, b1, acc01);
            acc10 = WMMA_F16(a1, b0, acc10);
            acc11 = WMMA_F16(a1, b1, acc11);
        }
        __syncthreads();

        if (wave == 0 && s + 2 < nsteps) {
            const _Float16* An = Ab + (size_t)(s + 2) * 64;
            const _Float16* Wn = Wb + (size_t)(s + 2) * 64;
            tdm_load_tile(An, (unsigned)(uintptr_t)&ldsA[s & 1][0], (unsigned)Mdim,
                          (unsigned)Kdim, 64u);
            tdm_load_tile(Wn, (unsigned)(uintptr_t)&ldsB[s & 1][0], (unsigned)Ndim,
                          (unsigned)Kdim, 128u);
        }
    }
#else
    // Fallback: direct global-load fragments (round-1 pipeline).
    const _Float16* Abase = A + (size_t)(mBlk + wm * 32) * Kdim;
    const _Float16* Wbase = W + (size_t)(nBlk + wn * 32) * Kdim;
    for (int k0 = 0; k0 < Kdim; k0 += 32) {
        const v16h a0 = load_frag16x32(Abase + k0,                     Kdim, lane);
        const v16h a1 = load_frag16x32(Abase + (size_t)16 * Kdim + k0, Kdim, lane);
        const v16h b0 = load_frag16x32(Wbase + k0,                     Kdim, lane);
        const v16h b1 = load_frag16x32(Wbase + (size_t)16 * Kdim + k0, Kdim, lane);
        acc00 = WMMA_F16(a0, b0, acc00);
        acc01 = WMMA_F16(a0, b1, acc01);
        acc10 = WMMA_F16(a1, b0, acc10);
        acc11 = WMMA_F16(a1, b1, acc11);
    }
#endif

    const int m0 = mBlk + wm * 32;
    const int n0 = nBlk + wn * 32;
    float* cbase = Out + (size_t)m0 * Ndim + n0;
    store_tile16x16(acc00, cbase,                          Ndim, lane, act);
    store_tile16x16(acc01, cbase + 16,                     Ndim, lane, act);
    store_tile16x16(acc10, cbase + (size_t)16 * Ndim,      Ndim, lane, act);
    store_tile16x16(acc11, cbase + (size_t)16 * Ndim + 16, Ndim, lane, act);
}

// ---------------------------------------------------------------------------
// WKV chunk-parallel scan.  State (sa, sb) obeys s' = ew*s + in  (linear!).
// Pass 1: per (b,h,chunk) wave: local decayed sums over LCH steps.
// Pass 2: per (b,h) wave: 32-step prefix combine with ew^L = exp(-L e^w).
// Pass 3: per (b,h,chunk) wave: replay chunk from true start state, emit
//         f16(r * wkv) for the output GEMM.
// One wave32 per chunk; lane owns channels 2*lane, 2*lane+1 (float2 loads);
// head-wide sums via 5-step shfl_xor butterfly.
// ---------------------------------------------------------------------------
__device__ __forceinline__ float wave_sum32(float x) {
#pragma unroll
    for (int m = 16; m >= 1; m >>= 1) x += __shfl_xor(x, m, 32);
    return x;
}

__global__ __launch_bounds__(256) void wkv_pass1_kernel(
    const float* __restrict__ k, const float* __restrict__ v,
    const float* __restrict__ time_decay,
    float* __restrict__ saLoc, float* __restrict__ sbLoc) {
    const int wgl  = blockIdx.x * (blockDim.x >> 5) + (threadIdx.x >> 5);
    const int lane = threadIdx.x & 31;
    if (wgl >= RB * RH * NCH) return;
    const int c  = wgl % NCH;
    const int bh = wgl / NCH;
    const int b  = bh / RH, h = bh % RH;

    const float ew = __expf(-__expf(time_decay[h]));
    const size_t base = ((size_t)(b * RT + c * LCH)) * RC + h * RD + 2 * lane;

    float a0 = 0.0f, a1 = 0.0f, sb = 0.0f;
    for (int i = 0; i < LCH; ++i) {
        const size_t off = base + (size_t)i * RC;
        const float2 kt = *reinterpret_cast<const float2*>(k + off);
        const float2 vt = *reinterpret_cast<const float2*>(v + off);
        const float e0 = __expf(kt.x), e1 = __expf(kt.y);
        const float es = wave_sum32(e0 + e1);
        a0 = ew * a0 + e0 * vt.x;
        a1 = ew * a1 + e1 * vt.y;
        sb = ew * sb + es;
    }
    saLoc[(size_t)wgl * RD + 2 * lane]     = a0;
    saLoc[(size_t)wgl * RD + 2 * lane + 1] = a1;
    if (lane == 0) sbLoc[wgl] = sb;
}

__global__ __launch_bounds__(256) void wkv_pass2_kernel(
    const float* __restrict__ saLoc, const float* __restrict__ sbLoc,
    const float* __restrict__ time_decay,
    float* __restrict__ saSt, float* __restrict__ sbSt) {
    const int bh   = blockIdx.x * (blockDim.x >> 5) + (threadIdx.x >> 5);
    const int lane = threadIdx.x & 31;
    if (bh >= RB * RH) return;
    const int h = bh % RH;
    const float ewL = __expf(-(float)LCH * __expf(time_decay[h]));  // ew^LCH exactly

    float S0 = 0.0f, S1 = 0.0f, Sb = 0.0f;
    for (int c = 0; c < NCH; ++c) {
        const size_t idx = (size_t)bh * NCH + c;
        saSt[idx * RD + 2 * lane]     = S0;       // state BEFORE chunk c
        saSt[idx * RD + 2 * lane + 1] = S1;
        if (lane == 0) sbSt[idx] = Sb;
        S0 = ewL * S0 + saLoc[idx * RD + 2 * lane];
        S1 = ewL * S1 + saLoc[idx * RD + 2 * lane + 1];
        Sb = ewL * Sb + sbLoc[idx];
    }
}

__global__ __launch_bounds__(256) void wkv_pass3_kernel(
    const float* __restrict__ k, const float* __restrict__ v,
    const float* __restrict__ r,
    const float* __restrict__ time_decay, const float* __restrict__ time_first,
    const float* __restrict__ saSt, const float* __restrict__ sbSt,
    _Float16* __restrict__ rwkv) {
    const int wgl  = blockIdx.x * (blockDim.x >> 5) + (threadIdx.x >> 5);
    const int lane = threadIdx.x & 31;
    if (wgl >= RB * RH * NCH) return;
    const int c  = wgl % NCH;
    const int bh = wgl / NCH;
    const int b  = bh / RH, h = bh % RH;

    const float ew = __expf(-__expf(time_decay[h]));
    const float u  = time_first[h];
    const size_t base = ((size_t)(b * RT + c * LCH)) * RC + h * RD + 2 * lane;

    float sa0 = saSt[(size_t)wgl * RD + 2 * lane];
    float sa1 = saSt[(size_t)wgl * RD + 2 * lane + 1];
    float sb  = sbSt[wgl];

    for (int i = 0; i < LCH; ++i) {
        const size_t off = base + (size_t)i * RC;
        const float2 kt = *reinterpret_cast<const float2*>(k + off);
        const float2 vt = *reinterpret_cast<const float2*>(v + off);
        const float2 rt = *reinterpret_cast<const float2*>(r + off);

        const float eu0 = __expf(u + kt.x), eu1 = __expf(u + kt.y);
        const float ek0 = __expf(kt.x),     ek1 = __expf(kt.y);
        const float eus = wave_sum32(eu0 + eu1);
        const float eks = wave_sum32(ek0 + ek1);

        const float rden = 1.0f / (sb + fmaxf(eus, 1e-6f));
        const float num0 = sa0 + eu0 * vt.x;
        const float num1 = sa1 + eu1 * vt.y;

        union { _Float16 h2[2]; unsigned int u32; } o;
        o.h2[0] = (_Float16)(rt.x * num0 * rden);
        o.h2[1] = (_Float16)(rt.y * num1 * rden);
        *reinterpret_cast<unsigned int*>(rwkv + off) = o.u32;

        sa0 = ew * sa0 + ek0 * vt.x;
        sa1 = ew * sa1 + ek1 * vt.y;
        sb  = ew * sb + eks;
    }
}

// ---------------------------------------------------------------------------
// Host launcher
// ---------------------------------------------------------------------------
extern "C" void kernel_launch(void* const* d_in, const int* in_sizes, int n_in,
                              void* d_out, int out_size, void* d_ws, size_t ws_size,
                              hipStream_t stream) {
    const float* x          = (const float*)d_in[0];
    const float* mix_k      = (const float*)d_in[1];
    const float* mix_v      = (const float*)d_in[2];
    const float* mix_r      = (const float*)d_in[3];
    const float* Wk         = (const float*)d_in[4];
    const float* Wv         = (const float*)d_in[5];
    const float* Wr         = (const float*)d_in[6];
    const float* Wo         = (const float*)d_in[7];
    const float* time_decay = (const float*)d_in[8];
    const float* time_first = (const float*)d_in[9];
    float* out = (float*)d_out;

    char* ws = (char*)d_ws;
    const size_t MC_h = (size_t)RM * RC * sizeof(_Float16);   // 16 MB
    const size_t CC_h = (size_t)RC * RC * sizeof(_Float16);   //  2 MB
    const size_t MC_f = (size_t)RM * RC * sizeof(float);      // 32 MB
    const size_t ST_f = (size_t)RB * RH * NCH * RD * sizeof(float);  // 512 KB
    const size_t SB_f = (size_t)RB * RH * NCH * sizeof(float);       //   8 KB

    size_t o = 0;
    _Float16* xk_h = (_Float16*)(ws + o); o += MC_h;          // reused for r*wkv
    _Float16* xv_h = (_Float16*)(ws + o); o += MC_h;
    _Float16* xr_h = (_Float16*)(ws + o); o += MC_h;
    _Float16* Wk_h = (_Float16*)(ws + o); o += CC_h;
    _Float16* Wv_h = (_Float16*)(ws + o); o += CC_h;
    _Float16* Wr_h = (_Float16*)(ws + o); o += CC_h;
    _Float16* Wo_h = (_Float16*)(ws + o); o += CC_h;
    float*    k_f  = (float*)(ws + o);    o += MC_f;
    float*    v_f  = (float*)(ws + o);    o += MC_f;
    float*    r_f  = (float*)(ws + o);    o += MC_f;
    float*    saLoc = (float*)(ws + o);   o += ST_f;
    float*    saSt  = (float*)(ws + o);   o += ST_f;
    float*    sbLoc = (float*)(ws + o);   o += SB_f;
    float*    sbSt  = (float*)(ws + o);   o += SB_f;
    _Float16* rwkv_h = xk_h;   // xk consumed by its GEMM before the scan runs

    // 1) token-shift mixes -> f16
    {
        const int n = RM * RC;
        mix_kernel<<<(n + 255) / 256, 256, 0, stream>>>(x, mix_k, mix_v, mix_r,
                                                        xk_h, xv_h, xr_h);
    }
    // 2) weight conversion -> f16
    {
        const int n = RC * RC;
        const int g = (n + 255) / 256;
        f32_to_f16_kernel<<<g, 256, 0, stream>>>(Wk, Wk_h, n);
        f32_to_f16_kernel<<<g, 256, 0, stream>>>(Wv, Wv_h, n);
        f32_to_f16_kernel<<<g, 256, 0, stream>>>(Wr, Wr_h, n);
        f32_to_f16_kernel<<<g, 256, 0, stream>>>(Wo, Wo_h, n);
    }
    // 3) k / v / r GEMMs (r fused with sigmoid)
    {
        dim3 grid(RC / 128, RM / 64);
        gemm_wmma_kernel<<<grid, 256, 0, stream>>>(xk_h, Wk_h, k_f, RM, RC, RC, 0);
        gemm_wmma_kernel<<<grid, 256, 0, stream>>>(xv_h, Wv_h, v_f, RM, RC, RC, 0);
        gemm_wmma_kernel<<<grid, 256, 0, stream>>>(xr_h, Wr_h, r_f, RM, RC, RC, 1);
    }
    // 4) chunk-parallel WKV scan (+ fused r*wkv -> f16)
    {
        const int waves1 = RB * RH * NCH;   // 2048 chunk waves
        const int blocks1 = (waves1 * 32 + 255) / 256;
        wkv_pass1_kernel<<<blocks1, 256, 0, stream>>>(k_f, v_f, time_decay,
                                                      saLoc, sbLoc);
        const int waves2 = RB * RH;         // 64 prefix waves
        const int blocks2 = (waves2 * 32 + 255) / 256;
        wkv_pass2_kernel<<<blocks2, 256, 0, stream>>>(saLoc, sbLoc, time_decay,
                                                      saSt, sbSt);
        wkv_pass3_kernel<<<blocks1, 256, 0, stream>>>(k_f, v_f, r_f,
                                                      time_decay, time_first,
                                                      saSt, sbSt, rwkv_h);
    }
    // 5) output GEMM
    {
        dim3 grid(RC / 128, RM / 64);
        gemm_wmma_kernel<<<grid, 256, 0, stream>>>(rwkv_h, Wo_h, out, RM, RC, RC, 0);
    }
}
